// Decouple_18442589569603
// MI455X (gfx1250) — matrix operand order repaired and checked by reading, not compiled, not run on hardware
//
#include <hip/hip_runtime.h>
#include <hip/hip_bf16.h>

// ---------------------------------------------------------------------------
// Decouple module for MI455X (gfx1250, wave32, WMMA).
//   out = (x1, m, e)
//   t  = conv3x3(x1,w_off)+b - 1x1(x1, sum w_off)      -> WMMA im2col GEMM
//        (im2col diff panel staged block-cooperatively in LDS)
//   mt = conv3x3(t, w_s1); et = conv3x3([p2,-p1], w_s2) -> VALU
//   m/e = 1x1( deform_dw(x1, mt/et, w1) - x1*sum(w1), w2 )
//         -> fused kernel: bilinear sampling into LDS f16 tile + WMMA GEMM
// ---------------------------------------------------------------------------

typedef __attribute__((ext_vector_type(16))) _Float16 v16h;
typedef __attribute__((ext_vector_type(8)))  float    v8f;

#define BATCH 4
#define CH    64
#define HH    128
#define WW    128
#define HWSZ  (HH * WW)        // 16384
#define NPIX  (BATCH * HWSZ)   // 65536
#define OFFC  18
#define KK    9                // 3x3 taps
#define KOFF  (CH * KK)        // 576 = im2col K for the offset conv
#define BPST  36               // padded panel stride (halves): 18-dword rows

__device__ __forceinline__ float sampleTap(const float* __restrict__ xc, int y, int x) {
    return (y >= 0 && y < HH && x >= 0 && x < WW) ? xc[y * WW + x] : 0.f;
}

// ---------------------------------------------------------------------------
// Kernel 1: t[b,oc,h,w] = b_off[oc]
//            + sum_{ic,ky,kx} w_off[oc,ic,ky,kx]*(x_shift - x_center)
// GEMM: M=18 (2 tiles of 16), K=576, N=65536. Block = one image row (128 px).
// Per K-step the 32x128 f16 im2col diff panel is built cooperatively in LDS:
// one k per thread (single k/9 decomposition), 16 coalesced pixels each.
// ---------------------------------------------------------------------------
__global__ __launch_bounds__(256)
void offset_field_kernel(const float* __restrict__ x1,
                         const float* __restrict__ w_off,   // [18][576]
                         const float* __restrict__ b_off,   // [18]
                         float* __restrict__ t)             // [B][18][H][W]
{
    __shared__ _Float16 wlds[32 * KOFF];     // weights, rows 18..31 zero (36 KB)
    __shared__ _Float16 bpanel[128 * BPST];  // [px][k_local] diff panel (9 KB)

    const int tid = threadIdx.x;
    for (int i = tid; i < 32 * KOFF; i += 256) {
        int row = i / KOFF, colk = i - row * KOFF;
        wlds[i] = (_Float16)(row < OFFC ? w_off[row * KOFF + colk] : 0.f);
    }

    const int n0  = blockIdx.x * 128;        // row-aligned (128 == W)
    const int b   = n0 >> 14;
    const int hw0 = n0 & (HWSZ - 1);
    const int h   = hw0 >> 7;
    const float* xb = x1 + b * CH * HWSZ;

    __builtin_prefetch(xb + hw0, 0, 0);      // global_prefetch_b8 into L2

    // panel-build mapping: one k_local per thread, 16 consecutive pixels
    const int kl  = tid >> 3;                // 0..31
    const int pxc = (tid & 7) * 16;          // pixel chunk base

    // wmma mapping
    const int wave = tid >> 5;
    const int lane = tid & 31;
    const int col  = lane & 15;              // N within 16-wide tile
    const int half = lane >> 4;
    const int p0   = wave * 16;              // wave's pixel base in the row
    const int kbA  = half * 16;

    v8f acc0 = {}, acc1 = {};

    for (int ks = 0; ks < KOFF; ks += 32) {
        __syncthreads();   // panel reuse WAR (1st iter: wlds RAW)

        // ---- build 32xK panel slice: bpanel[px][kl] = x_shift - x_center
        {
            int k  = ks + kl;
            int ic = k / KK;                 // once per thread per step
            int s  = k - ic * KK;
            int ky = s / 3, kx = s - ky * 3;
            int yy = h + ky - 1;
            const float* xc = xb + ic * HWSZ;
            const float* xr = xc + yy * WW;
            const bool rowok = (yy >= 0) && (yy < HH);
            #pragma unroll
            for (int p = 0; p < 16; ++p) {
                int px = pxc + p;
                int xx = px + kx - 1;        // coalesced row access
                float v = (rowok && xx >= 0 && xx < WW) ? xr[xx] : 0.f;
                bpanel[px * BPST + kl] = (_Float16)(v - xc[hw0 + px]);
            }
        }
        __syncthreads();

        // ---- B fragment: 16 contiguous halves per lane -> ds_load_b128 x2
        v16h bfrag;
        #pragma unroll
        for (int j = 0; j < 16; ++j)
            bfrag[j] = bpanel[(p0 + col) * BPST + half * 16 + j];

        // ---- A fragments (16x32 f16) from LDS
        v16h a0, a1;
        #pragma unroll
        for (int j = 0; j < 8; ++j) {
            a0[j]     = wlds[col        * KOFF + ks + kbA + j];
            a0[j + 8] = wlds[col        * KOFF + ks + kbA + 8 + j];
            a1[j]     = wlds[(col + 16) * KOFF + ks + kbA + j];
            a1[j + 8] = wlds[(col + 16) * KOFF + ks + kbA + 8 + j];
        }
        acc0 = __builtin_amdgcn_wmma_f32_16x16x32_f16(false, a0, false, bfrag,
                                                      (short)0, acc0, false, false);
        acc1 = __builtin_amdgcn_wmma_f32_16x16x32_f16(false, a1, false, bfrag,
                                                      (short)0, acc1, false, false);
    }

    // D layout: lanes0-15 VGPR r -> M=r ; lanes16-31 -> M=8+r ; N=col
    const int hw = hw0 + p0 + col;
    #pragma unroll
    for (int r = 0; r < 8; ++r) {
        int oc0 = half * 8 + r;
        t[(b * OFFC + oc0) * HWSZ + hw] = acc0[r] + b_off[oc0];
        int oc1 = oc0 + 16;
        if (oc1 < OFFC)
            t[(b * OFFC + oc1) * HWSZ + hw] = acc1[r] + b_off[oc1];
    }
}

// ---------------------------------------------------------------------------
// Kernel 2: mt = conv3x3(t, w_s1, pad=1) ; et = conv3x3([p2,-p1], w_s2, pad=1)
// which/oc uniform per block -> weights in LDS.
// ---------------------------------------------------------------------------
__global__ __launch_bounds__(256)
void spatial_conv_kernel(const float* __restrict__ t,     // [B][18][H][W]
                         const float* __restrict__ w_s1,  // [18][18][3][3]
                         const float* __restrict__ w_s2,
                         float* __restrict__ mt,
                         float* __restrict__ et)
{
    __shared__ float wsl[OFFC * KK];     // 162 weights for this (which, oc)

    const int gid   = blockIdx.x * 256 + threadIdx.x;
    const int which = gid / (OFFC * NPIX);        // uniform per block
    const int rem   = gid - which * OFFC * NPIX;
    const int oc    = rem / NPIX;                 // uniform per block
    const int n     = rem - oc * NPIX;

    const float* wsel = which ? w_s2 : w_s1;
    if (threadIdx.x < OFFC * KK)
        wsl[threadIdx.x] = wsel[oc * OFFC * KK + threadIdx.x];
    __syncthreads();

    const int b = n >> 14, hw = n & (HWSZ - 1);
    const int h = hw >> 7, w = hw & (WW - 1);
    const float* tb = t + b * OFFC * HWSZ;

    float acc = 0.f;
    for (int ic = 0; ic < OFFC; ++ic) {
        // et input channel map: [p2, -p1]
        int   src = which ? (ic < 9 ? ic + 9 : ic - 9) : ic;
        float sgn = (which && ic >= 9) ? -1.f : 1.f;
        const float* tc = tb + src * HWSZ;
        #pragma unroll
        for (int s = 0; s < KK; ++s) {
            int ky = s / 3, kx = s - ky * 3;
            int yy = h + ky - 1, xx = w + kx - 1;
            if (yy >= 0 && yy < HH && xx >= 0 && xx < WW)
                acc += wsl[ic * KK + s] * sgn * tc[yy * WW + xx];
        }
    }
    (which ? et : mt)[(b * OFFC + oc) * HWSZ + hw] = acc;
}

// ---------------------------------------------------------------------------
// Kernel 3 (fused traj_encode): one block = one image row (128 pixels).
//  phase 0: cache offsets (9 taps/pixel) + weights in LDS
//  phase 1: deformable depthwise sampling + diff -> 64x128 f16 LDS tile
//  phase 2: 1x1 conv as WMMA GEMM (64x64 weights, f16) straight from LDS
// ---------------------------------------------------------------------------
__global__ __launch_bounds__(256)
void traj_encode_kernel(const float* __restrict__ x1,
                        const float* __restrict__ off,  // [B][18][H][W]
                        const float* __restrict__ w1,   // [64][9] depthwise
                        const float* __restrict__ w2,   // [64][64] pointwise
                        float* __restrict__ out)        // [B][64][H][W]
{
    __shared__ float    pys[128 * KK];
    __shared__ float    pxs[128 * KK];
    __shared__ float    w1s[CH * KK];
    __shared__ float    wsum[CH];
    __shared__ _Float16 w2h[CH * CH];
    __shared__ _Float16 ylds[CH][128];

    const int tid = threadIdx.x;
    const int n0  = blockIdx.x * 128;          // row-aligned (128 == W)
    const int b   = n0 >> 14;
    const int hw0 = n0 & (HWSZ - 1);
    const int h   = hw0 >> 7;

    for (int i = tid; i < CH * KK; i += 256) w1s[i] = w1[i];
    for (int i = tid; i < CH * CH; i += 256) w2h[i] = (_Float16)w2[i];
    __syncthreads();

    if (tid < CH) {
        float s = 0.f;
        #pragma unroll
        for (int k = 0; k < KK; ++k) s += w1s[tid * KK + k];
        wsum[tid] = s;
    }
    const float* ob = off + b * OFFC * HWSZ;
    for (int i = tid; i < 128 * KK; i += 256) {
        int px = i / KK, k = i - px * KK;
        int ky = k / 3, kx = k - ky * 3;
        float dy = ob[(2 * k)     * HWSZ + hw0 + px];
        float dx = ob[(2 * k + 1) * HWSZ + hw0 + px];
        pys[i] = (float)(h + ky - 1) + dy;
        pxs[i] = (float)(px + kx - 1) + dx;
    }
    __syncthreads();

    // phase 1: y[c][w] = sum_k w1[c,k]*bilinear(x1[b,c]) - x1[b,c,h,w]*wsum[c]
    for (int i = tid; i < CH * 128; i += 256) {
        int c = i >> 7, w = i & 127;
        const float* xc = x1 + (b * CH + c) * HWSZ;
        float acc = 0.f;
        #pragma unroll
        for (int k = 0; k < KK; ++k) {
            float py = pys[w * KK + k], px = pxs[w * KK + k];
            float fy = floorf(py), fx = floorf(px);
            int   y0 = (int)fy,    x0 = (int)fx;
            float wy = py - fy,    wx = px - fx;
            float v00 = sampleTap(xc, y0,     x0);
            float v01 = sampleTap(xc, y0,     x0 + 1);
            float v10 = sampleTap(xc, y0 + 1, x0);
            float v11 = sampleTap(xc, y0 + 1, x0 + 1);
            float sv  = v00 * (1.f - wy) * (1.f - wx) + v01 * (1.f - wy) * wx
                      + v10 * wy * (1.f - wx)         + v11 * wy * wx;
            acc += w1s[c * KK + k] * sv;
        }
        acc -= xc[hw0 + w] * wsum[c];
        ylds[c][w] = (_Float16)acc;
    }
    __syncthreads();

    // phase 2: D[64x16 per wave] = W2[64x64] x Y[64x16], K-steps of 32
    const int wave = tid >> 5;
    const int lane = tid & 31;
    const int col  = lane & 15;
    const int half = lane >> 4;
    const int p0   = wave * 16;
    v8f acc[4] = {};

    #pragma unroll
    for (int ks = 0; ks < CH; ks += 32) {
        v16h bfrag;
        #pragma unroll
        for (int j = 0; j < 16; ++j)
            bfrag[j] = ylds[ks + half * 16 + j][p0 + col];
        #pragma unroll
        for (int mt = 0; mt < 4; ++mt) {
            v16h a;
            int m = mt * 16 + col;
            #pragma unroll
            for (int j = 0; j < 8; ++j) {
                a[j]     = w2h[m * CH + ks + half * 16 + j];
                a[j + 8] = w2h[m * CH + ks + half * 16 + 8 + j];
            }
            acc[mt] = __builtin_amdgcn_wmma_f32_16x16x32_f16(false, a, false, bfrag,
                                                             (short)0, acc[mt], false, false);
        }
    }
    #pragma unroll
    for (int mt = 0; mt < 4; ++mt) {
        #pragma unroll
        for (int r = 0; r < 8; ++r) {
            int oc = mt * 16 + half * 8 + r;
            out[(b * CH + oc) * HWSZ + hw0 + p0 + col] = acc[mt][r];
        }
    }
}

// ---------------------------------------------------------------------------
extern "C" void kernel_launch(void* const* d_in, const int* in_sizes, int n_in,
                              void* d_out, int out_size, void* d_ws, size_t ws_size,
                              hipStream_t stream) {
    const float* x1    = (const float*)d_in[0];
    const float* w_off = (const float*)d_in[1];
    const float* b_off = (const float*)d_in[2];
    const float* w_s1  = (const float*)d_in[3];
    const float* w_s2  = (const float*)d_in[4];
    const float* w_m1  = (const float*)d_in[5];
    const float* w_m2  = (const float*)d_in[6];
    const float* w_e1  = (const float*)d_in[7];
    const float* w_e2  = (const float*)d_in[8];

    float* t_buf  = (float*)d_ws;              // 18*65536 f32
    float* mt_buf = t_buf  + OFFC * NPIX;
    float* et_buf = mt_buf + OFFC * NPIX;      // total ws use: 14.2 MB

    float* out_x = (float*)d_out;
    float* out_m = out_x + CH * NPIX;
    float* out_e = out_m + CH * NPIX;

    // output[0] = x1 passthrough
    hipMemcpyAsync(out_x, x1, sizeof(float) * (size_t)CH * NPIX,
                   hipMemcpyDeviceToDevice, stream);

    offset_field_kernel<<<NPIX / 128, 256, 0, stream>>>(x1, w_off, b_off, t_buf);
    spatial_conv_kernel<<<2 * OFFC * NPIX / 256, 256, 0, stream>>>(t_buf, w_s1, w_s2,
                                                                   mt_buf, et_buf);
    traj_encode_kernel<<<NPIX / 128, 256, 0, stream>>>(x1, mt_buf, w_m1, w_m2, out_m);
    traj_encode_kernel<<<NPIX / 128, 256, 0, stream>>>(x1, et_buf, w_e1, w_e2, out_e);
}
